// MOE_54125177864812
// MI455X (gfx1250) — matrix operand order repaired and checked by reading, not compiled, not run on hardware
//
#include <hip/hip_runtime.h>
#include <hip/hip_bf16.h>
#include <stdint.h>

#define E_ 5
#define H_ 768
#define I_ 1024
#define T_ 8192

typedef __bf16 bf16;
typedef bf16 bf16x16 __attribute__((ext_vector_type(16)));
typedef float f32x8 __attribute__((ext_vector_type(8)));

union FragU {
    bf16x16 v;
    uint4   q[2];
};

__device__ __forceinline__ f32x8 wmma_bf16(bf16x16 a, bf16x16 b, f32x8 c) {
    return __builtin_amdgcn_wmma_f32_16x16x32_bf16(
        /*neg_a=*/false, a, /*neg_b=*/false, b,
        /*c_mod=*/(short)0, c, /*reuse_a=*/false, /*reuse_b=*/false);
}

// ---------------------------------------------------------------------------
// Kernel 1: fp32 -> bf16 elementwise (x rows, row-major = WMMA A layout)
// ---------------------------------------------------------------------------
__global__ void cvt_x_kernel(const float* __restrict__ x, bf16* __restrict__ xb, int n) {
    int i = blockIdx.x * blockDim.x + threadIdx.x;
    if (i < n) xb[i] = (bf16)x[i];
}

// ---------------------------------------------------------------------------
// Kernel 2: tiled transpose + fp32->bf16 per expert slice.
// src: [E][R][C] fp32 row-major  ->  dst: [E][C][R] bf16 row-major (= B^T)
// R, C multiples of 32. block (32,8), grid (C/32, R/32, E)
// ---------------------------------------------------------------------------
__global__ __launch_bounds__(256)
void transpose_convert(const float* __restrict__ src, bf16* __restrict__ dst,
                       int R, int C) {
    __shared__ float tile[32][33];
    const float* s = src + (size_t)blockIdx.z * R * C;
    bf16*        d = dst + (size_t)blockIdx.z * R * C;
    int c0 = blockIdx.x * 32, r0 = blockIdx.y * 32;
#pragma unroll
    for (int i = threadIdx.y; i < 32; i += 8) {
        tile[i][threadIdx.x] = s[(size_t)(r0 + i) * C + c0 + threadIdx.x];
    }
    __syncthreads();
#pragma unroll
    for (int i = threadIdx.y; i < 32; i += 8) {
        d[(size_t)(c0 + i) * R + r0 + threadIdx.x] = (bf16)tile[threadIdx.x][i];
    }
}

// ---------------------------------------------------------------------------
// Kernel 3: router. One wave per token. logits -> softmax -> top2 -> renorm,
// append (token, weight) into per-expert buckets with atomics.
// block 256 (8 waves), grid T/8
// ---------------------------------------------------------------------------
__global__ __launch_bounds__(256)
void router_kernel(const float* __restrict__ x, const float* __restrict__ rw,
                   const float* __restrict__ rb, int* __restrict__ cnt,
                   int* __restrict__ btok, float* __restrict__ bwgt) {
    int wave = threadIdx.x >> 5, lane = threadIdx.x & 31;
    int t = blockIdx.x * 8 + wave;
    float p[E_] = {0.f, 0.f, 0.f, 0.f, 0.f};
    const float* xr = x + (size_t)t * H_;
    for (int h = lane; h < H_; h += 32) {
        float xv = xr[h];
#pragma unroll
        for (int e = 0; e < E_; ++e) p[e] += xv * rw[e * H_ + h];
    }
#pragma unroll
    for (int e = 0; e < E_; ++e)
        for (int off = 16; off > 0; off >>= 1) p[e] += __shfl_xor(p[e], off, 32);
    if (lane == 0) {
        float lg[E_], ex[E_];
        float mx = -3.4e38f;
#pragma unroll
        for (int e = 0; e < E_; ++e) { lg[e] = p[e] + rb[e]; mx = fmaxf(mx, lg[e]); }
#pragma unroll
        for (int e = 0; e < E_; ++e) ex[e] = __expf(lg[e] - mx);
        // top-2 on ex (monotone in scores); softmax denom cancels in renorm
        int i0 = 0;
#pragma unroll
        for (int e = 1; e < E_; ++e) if (ex[e] > ex[i0]) i0 = e;
        int i1 = (i0 == 0) ? 1 : 0;
#pragma unroll
        for (int e = 0; e < E_; ++e) if (e != i1 && e != i0 && ex[e] > ex[i1]) i1 = e;
        float denom = ex[i0] + ex[i1];
        int   ei[2] = {i0, i1};
        float wv[2] = {ex[i0] / denom, ex[i1] / denom};
#pragma unroll
        for (int k = 0; k < 2; ++k) {
            int e   = ei[k];
            int pos = atomicAdd(&cnt[e], 1);
            btok[e * T_ + pos] = t;
            bwgt[e * T_ + pos] = wv[k];
        }
    }
}

// ---------------------------------------------------------------------------
// Kernel 4: fused expert SwiGLU + down-proj, WMMA bf16, M-tile = 32 tokens.
// block 512 (16 waves = 2 M-subtiles x 8 N-chunks); grid (T/32, E);
// blocks beyond ceil(cnt[e]/32) exit.
// Phase 1: wave (mi,nc) -> rows [16*mi,16*mi+16), cols [128*nc,128*nc+128) of I:
//          g,u accumulators, K over H; h = silu(g)*u -> LDS (32 x 1024 bf16).
// Phase 2: wave (mi,nc) -> cols [96*nc, 96*nc+96) of H, K over I,
//          weighted atomicAdd scatter into out.
// ---------------------------------------------------------------------------
#define M_TILE 32
#define XS_STRIDE (H_ + 8)   // bf16 elems; row stride 1552 B (16B aligned)
#define HS_STRIDE (I_ + 8)   // bf16 elems; row stride 2064 B (16B aligned)

__global__ __launch_bounds__(512)
void moe_expert_kernel(const bf16* __restrict__ xb,
                       const bf16* __restrict__ gwt,   // [E][I][H] = gate^T
                       const bf16* __restrict__ uwt,   // [E][I][H] = up^T
                       const bf16* __restrict__ dwt,   // [E][H][I] = down^T
                       const int* __restrict__ cnt,
                       const int* __restrict__ btok,
                       const float* __restrict__ bwgt,
                       float* __restrict__ out) {
    int e    = blockIdx.y;
    int tile = blockIdx.x;
    int ne   = cnt[e];
    if (tile * M_TILE >= ne) return;

    __shared__ __align__(16) bf16 xs[M_TILE * XS_STRIDE];
    __shared__ __align__(16) bf16 hs[M_TILE * HS_STRIDE];
    __shared__ int   tok_s[M_TILE];
    __shared__ float w_s[M_TILE];

    int tid = threadIdx.x;
    if (tid < M_TILE) {
        int  r     = tile * M_TILE + tid;
        bool valid = r < ne;
        int  src   = e * T_ + (valid ? r : 0);   // duplicate row 0 when padding
        tok_s[tid] = btok[src];
        w_s[tid]   = valid ? bwgt[src] : 0.0f;   // padded rows contribute 0
    }
    __syncthreads();

    // Gather 32 token rows (768 bf16 = 96 uint4 each) into LDS
    for (int q = tid; q < M_TILE * 96; q += 512) {
        int ri = q / 96, off = q % 96;
        const uint4* src = (const uint4*)(xb + (size_t)tok_s[ri] * H_) + off;
        ((uint4*)&xs[ri * XS_STRIDE])[off] = *src;
    }
    __syncthreads();

    int wv   = tid >> 5;
    int lane = tid & 31;
    int row  = lane & 15;    // A row / B col / C n-within-tile
    int half = lane >> 4;
    int mi   = wv & 1;       // which 16-row subtile
    int nc   = wv >> 1;      // which N chunk
    int mrow = mi * 16 + row;

    // ----------------- Phase 1: g = x@Wg, u = x@Wu, h = silu(g)*u ----------
    {
        f32x8 accg[8] = {};
        f32x8 accu[8] = {};
        const bf16* gb = gwt + (size_t)e * I_ * H_;
        const bf16* ub = uwt + (size_t)e * I_ * H_;
        int nbase = nc * 128;
        for (int k0 = 0; k0 < H_; k0 += 32) {
            int kA = k0 + half * 8;        // A frag: {k0+8h .. +7, k0+16+8h .. +7}
            int kB = k0 + 16 + half * 8;
            int kC = k0 + half * 8;        // B frag: same order as A
            FragU a;
            a.q[0] = *(const uint4*)&xs[mrow * XS_STRIDE + kA];
            a.q[1] = *(const uint4*)&xs[mrow * XS_STRIDE + kB];
#pragma unroll
            for (int nt = 0; nt < 8; ++nt) {
                int col = nbase + nt * 16 + row;
                const bf16* gp = gb + (size_t)col * H_;
                const bf16* up = ub + (size_t)col * H_;
                FragU bg, bu;
                bg.q[0] = *(const uint4*)&gp[kC];
                bg.q[1] = *(const uint4*)&gp[kC + 16];
                bu.q[0] = *(const uint4*)&up[kC];
                bu.q[1] = *(const uint4*)&up[kC + 16];
                accg[nt] = wmma_bf16(a.v, bg.v, accg[nt]);
                accu[nt] = wmma_bf16(a.v, bu.v, accu[nt]);
            }
        }
        // silu(g)*u -> LDS h tile (bf16). C layout: m=j+8*half, n=row+16*nt
#pragma unroll
        for (int nt = 0; nt < 8; ++nt) {
            int n = nbase + nt * 16 + row;
#pragma unroll
            for (int j = 0; j < 8; ++j) {
                int   m = mi * 16 + j + half * 8;
                float g = accg[nt][j];
                float h = (g / (1.0f + __expf(-g))) * accu[nt][j];
                hs[m * HS_STRIDE + n] = (bf16)h;
            }
        }
    }
    __syncthreads();

    // ----------------- Phase 2: y = h @ Wd, scatter w * y -------------------
    {
        f32x8 acc[6] = {};
        const bf16* db = dwt + (size_t)e * H_ * I_;
        int nbase = nc * 96;
        for (int k0 = 0; k0 < I_; k0 += 32) {
            int kA = k0 + half * 8;
            int kB = k0 + 16 + half * 8;
            int kC = k0 + half * 8;
            FragU a;
            a.q[0] = *(const uint4*)&hs[mrow * HS_STRIDE + kA];
            a.q[1] = *(const uint4*)&hs[mrow * HS_STRIDE + kB];
#pragma unroll
            for (int nt = 0; nt < 6; ++nt) {
                int col = nbase + nt * 16 + row;
                const bf16* dp = db + (size_t)col * I_;
                FragU bd;
                bd.q[0] = *(const uint4*)&dp[kC];
                bd.q[1] = *(const uint4*)&dp[kC + 16];
                acc[nt] = wmma_bf16(a.v, bd.v, acc[nt]);
            }
        }
#pragma unroll
        for (int nt = 0; nt < 6; ++nt) {
            int n = nbase + nt * 16 + row;
#pragma unroll
            for (int j = 0; j < 8; ++j) {
                int m = mi * 16 + j + half * 8;
                atomicAdd(out + (size_t)tok_s[m] * H_ + n, w_s[m] * acc[nt][j]);
            }
        }
    }
}

// ---------------------------------------------------------------------------
extern "C" void kernel_launch(void* const* d_in, const int* in_sizes, int n_in,
                              void* d_out, int out_size, void* d_ws, size_t ws_size,
                              hipStream_t stream) {
    (void)in_sizes; (void)n_in; (void)ws_size;
    const float* x        = (const float*)d_in[0];
    const float* router_w = (const float*)d_in[1];
    const float* router_b = (const float*)d_in[2];
    const float* gate_w   = (const float*)d_in[3];
    const float* up_w     = (const float*)d_in[4];
    const float* down_w   = (const float*)d_in[5];
    float*       out      = (float*)d_out;

    char*  ws  = (char*)d_ws;
    size_t off = 0;
    auto take = [&](size_t bytes) -> char* {
        char* p = ws + off;
        off = (off + bytes + 255) & ~(size_t)255;
        return p;
    };
    bf16*  xb   = (bf16*)take((size_t)T_ * H_ * 2);
    bf16*  gwt  = (bf16*)take((size_t)E_ * I_ * H_ * 2);
    bf16*  uwt  = (bf16*)take((size_t)E_ * I_ * H_ * 2);
    bf16*  dwt  = (bf16*)take((size_t)E_ * H_ * I_ * 2);
    int*   cnt  = (int*)take(256);
    int*   btok = (int*)take((size_t)E_ * T_ * 4);
    float* bwgt = (float*)take((size_t)E_ * T_ * 4);

    hipMemsetAsync(cnt, 0, 256, stream);
    hipMemsetAsync(out, 0, (size_t)out_size * sizeof(float), stream);

    int nx = T_ * H_;
    cvt_x_kernel<<<(nx + 255) / 256, 256, 0, stream>>>(x, xb, nx);

    dim3 tb(32, 8);
    transpose_convert<<<dim3(I_ / 32, H_ / 32, E_), tb, 0, stream>>>(gate_w, gwt, H_, I_);
    transpose_convert<<<dim3(I_ / 32, H_ / 32, E_), tb, 0, stream>>>(up_w,   uwt, H_, I_);
    transpose_convert<<<dim3(H_ / 32, I_ / 32, E_), tb, 0, stream>>>(down_w, dwt, I_, H_);

    router_kernel<<<T_ / 8, 256, 0, stream>>>(x, router_w, router_b, cnt, btok, bwgt);

    moe_expert_kernel<<<dim3(T_ / M_TILE, E_), 512, 0, stream>>>(
        xb, gwt, uwt, dwt, cnt, btok, bwgt, out);
}
